// VQ_9036611190897
// MI455X (gfx1250) — matrix operand order repaired and checked by reading, not compiled, not run on hardware
//
#include <hip/hip_runtime.h>

typedef __attribute__((ext_vector_type(16))) __bf16 bf16x16;
typedef __attribute__((ext_vector_type(8)))  __bf16 bf16x8;
typedef __attribute__((ext_vector_type(8)))  float  floatx8;

namespace {
constexpr int kD       = 256;    // embedding dim
constexpr int kCodes   = 1024;   // codebook size
constexpr int kRows    = 128;    // x rows per workgroup
constexpr int kThreads = 256;    // 8 waves (wave32)
constexpr int kStride  = 264;    // LDS row stride in bf16 elems (padded)
constexpr int kN       = 65536;  // total rows = 16*64*64
}

union AFrag { bf16x16 v; bf16x8 h[2]; };

// ---------------------------------------------------------------------------
// Prep: split codebook into bf16 hi/lo planes + per-code squared norms.
// ---------------------------------------------------------------------------
__global__ __launch_bounds__(256) void vq_prep(const float* __restrict__ w,
                                               __bf16* __restrict__ w_hi,
                                               __bf16* __restrict__ w_lo,
                                               float* __restrict__ w_sq) {
  const int code = blockIdx.x;          // one code row per block
  const int t    = threadIdx.x;         // t == d (256 threads)
  const float v = w[code * kD + t];
  const __bf16 h = (__bf16)v;
  w_hi[code * kD + t] = h;
  w_lo[code * kD + t] = (__bf16)(v - (float)h);

  float s = v * v;
  #pragma unroll
  for (int m = 16; m >= 1; m >>= 1) s += __shfl_xor(s, m, 32);
  __shared__ float red[8];
  const int lane = t & 31, wave = t >> 5;
  if (lane == 0) red[wave] = s;
  __syncthreads();
  if (t == 0) {
    float tot = 0.f;
    #pragma unroll
    for (int i = 0; i < 8; ++i) tot += red[i];
    w_sq[code] = tot;
  }
}

// ---------------------------------------------------------------------------
// Main: per-WG 128 rows; split-bf16 WMMA GEMM vs all codes, running argmin,
// gather-write quantized output, per-block squared-error partial.
// ---------------------------------------------------------------------------
__global__ __launch_bounds__(256) void vq_main(const float* __restrict__ x,
                                               const float* __restrict__ w,
                                               const __bf16* __restrict__ w_hi,
                                               const __bf16* __restrict__ w_lo,
                                               const float* __restrict__ w_sq,
                                               float* __restrict__ out_q,
                                               float* __restrict__ out_idx,
                                               float* __restrict__ partial) {
  __shared__ __attribute__((aligned(16))) __bf16 s_xhi[kRows * kStride];
  __shared__ __attribute__((aligned(16))) __bf16 s_xlo[kRows * kStride];
  __shared__ int   s_idx[kRows];
  __shared__ float s_red[8];

  const int tid  = threadIdx.x;
  const int lane = tid & 31;
  const int wave = tid >> 5;
  const int row0 = blockIdx.x * kRows;   // global row base (128 | 4096)
  const int bB   = row0 >> 12;           // batch index
  const int hw0  = row0 & 4095;          // hw offset within batch
  const float* xbase = x + ((size_t)bB << 20) + hw0;   // b*256*4096

  // ---- Stage + split x tile: coalesced NCHW read -> LDS [row][d] ----
  for (int i = tid; i < kRows * kD; i += kThreads) {
    const int r = i & (kRows - 1);
    const int d = i >> 7;
    const float v = xbase[((size_t)d << 12) + r];
    const __bf16 h = (__bf16)v;
    s_xhi[r * kStride + d] = h;
    s_xlo[r * kStride + d] = (__bf16)(v - (float)h);
  }
  __syncthreads();

  const int mrow  = wave * 16;     // this wave's 16-row group
  const int khalf = lane >> 4;     // lane-half selects K sub-range
  const int ncol  = lane & 15;     // B column / A row within tile

  float best[8];
  int   bidx[8];
  #pragma unroll
  for (int v = 0; v < 8; ++v) { best[v] = 3.4e38f; bidx[v] = 0; }

  for (int nt = 0; nt < kCodes / 16; ++nt) {
    const int code = nt * 16 + ncol;
    floatx8 acc = {};
    #pragma unroll
    for (int kc = 0; kc < kD / 32; ++kc) {
      const int kb = kc * 32;
      // A fragment: lane half holds 16 contiguous K values (k = 16*khalf + s), matching B
      const __bf16* ah = &s_xhi[(mrow + ncol) * kStride + kb + khalf * 16];
      const __bf16* al = &s_xlo[(mrow + ncol) * kStride + kb + khalf * 16];
      AFrag a_hi, a_lo;
      a_hi.h[0] = *(const bf16x8*)(ah);
      a_hi.h[1] = *(const bf16x8*)(ah + 8);
      a_lo.h[0] = *(const bf16x8*)(al);
      a_lo.h[1] = *(const bf16x8*)(al + 8);
      // B fragment: lane ncol holds 16 contiguous K values of code row
      const bf16x16 b_hi = *(const bf16x16*)(w_hi + (size_t)code * kD + kb + khalf * 16);
      const bf16x16 b_lo = *(const bf16x16*)(w_lo + (size_t)code * kD + kb + khalf * 16);
      // split-bf16: dot ~= hi*hi + hi*lo + lo*hi  (fp32 accumulate)
      acc = __builtin_amdgcn_wmma_f32_16x16x32_bf16(false, a_hi.v, false, b_hi, (short)0, acc, false, false);
      acc = __builtin_amdgcn_wmma_f32_16x16x32_bf16(false, a_hi.v, false, b_lo, (short)0, acc, false, false);
      acc = __builtin_amdgcn_wmma_f32_16x16x32_bf16(false, a_lo.v, false, b_hi, (short)0, acc, false, false);
    }
    // score = ||w||^2 - 2 x.w  (||x||^2 constant per row; irrelevant to argmin)
    const float wsq = w_sq[code];
    #pragma unroll
    for (int v = 0; v < 8; ++v) {
      const float s = fmaf(-2.f, acc[v], wsq);
      if (s < best[v]) { best[v] = s; bidx[v] = code; }  // strict < keeps first min
    }
  }

  // ---- Cross-lane argmin over the 16 columns of each C half ----
  #pragma unroll
  for (int v = 0; v < 8; ++v) {
    float bs = best[v]; int bi = bidx[v];
    #pragma unroll
    for (int m = 8; m >= 1; m >>= 1) {
      const float os = __shfl_xor(bs, m, 32);
      const int   oi = __shfl_xor(bi, m, 32);
      if (os < bs || (os == bs && oi < bi)) { bs = os; bi = oi; }
    }
    bidx[v] = bi;
  }
  if (ncol == 0) {                 // lanes 0 and 16 hold rows v and 8+v
    const int mb = mrow + khalf * 8;
    #pragma unroll
    for (int v = 0; v < 8; ++v) s_idx[mb + v] = bidx[v];
  }
  __syncthreads();

  // ---- Output pass: quantized = w[idx] (coalesced NCHW writes) + MSE ----
  float lsum = 0.f;
  for (int i = tid; i < kRows * kD; i += kThreads) {
    const int r = i & (kRows - 1);
    const int d = i >> 7;
    const int code = s_idx[r];
    const float q  = w[(size_t)code * kD + d];
    const size_t off = ((size_t)bB << 20) + ((size_t)d << 12) + (size_t)(hw0 + r);
    const float df = q - x[off];
    out_q[off] = q;
    lsum = fmaf(df, df, lsum);
  }
  if (tid < kRows) out_idx[row0 + tid] = (float)s_idx[tid];

  #pragma unroll
  for (int m = 16; m >= 1; m >>= 1) lsum += __shfl_xor(lsum, m, 32);
  if (lane == 0) s_red[wave] = lsum;
  __syncthreads();
  if (tid == 0) {
    float tot = 0.f;
    #pragma unroll
    for (int i = 0; i < 8; ++i) tot += s_red[i];
    partial[blockIdx.x] = tot;       // deterministic: no float atomics
  }
}

// ---------------------------------------------------------------------------
// Finalize: fixed-order sum of 512 block partials -> loss scalar.
// loss = q_latent + 0.25*e_latent = 1.25 * mean((q - x)^2)
// ---------------------------------------------------------------------------
__global__ void vq_finalize(const float* __restrict__ partial,
                            float* __restrict__ loss_out) {
  float tot = 0.f;
  for (int i = 0; i < kN / kRows; ++i) tot += partial[i];
  *loss_out = 1.25f * tot * (1.0f / 16777216.0f);   // N*D = 65536*256
}

// ---------------------------------------------------------------------------
extern "C" void kernel_launch(void* const* d_in, const int* in_sizes, int n_in,
                              void* d_out, int out_size, void* d_ws, size_t ws_size,
                              hipStream_t stream) {
  const float* x = (const float*)d_in[0];   // [16,256,64,64] fp32
  const float* w = (const float*)d_in[1];   // [1024,256] fp32
  float* out = (float*)d_out;               // quantized | indices | loss

  // workspace layout (bytes): w_hi 512K | w_lo 512K | w_sq 4K | partials 2K
  char* ws = (char*)d_ws;
  __bf16* w_hi  = (__bf16*)(ws);
  __bf16* w_lo  = (__bf16*)(ws + 524288);
  float*  w_sq  = (float*)(ws + 1048576);
  float*  parts = (float*)(ws + 1048576 + 4096);

  float* out_q    = out;                         // 16777216 elems
  float* out_idx  = out + 16777216;              // 65536 elems (float-cast ints)
  float* out_loss = out + 16777216 + 65536;      // 1 elem

  vq_prep<<<kCodes, 256, 0, stream>>>(w, w_hi, w_lo, w_sq);
  vq_main<<<kN / kRows, kThreads, 0, stream>>>(x, w, w_hi, w_lo, w_sq,
                                               out_q, out_idx, parts);
  vq_finalize<<<1, 1, 0, stream>>>(parts, out_loss);
}
